// inception_v1_52596169507108
// MI455X (gfx1250) — compile-verified, matrix operand order checked by reading, not run on hardware
//
#include <hip/hip_runtime.h>
#include <hip/hip_bf16.h>

typedef __attribute__((ext_vector_type(16))) _Float16 v16h;
typedef __attribute__((ext_vector_type(8)))  _Float16 v8h;
typedef __attribute__((ext_vector_type(8)))  float    v8f;

union V16J { v16h v; v8h h[2]; };

// ---------------- problem constants ----------------
#define TT 15
#define CIN 6
#define HIN 256
#define HXP 260           // conv1 input padded (pad=2 each side)
#define CXP 7             // 6 real channels + 1 zero channel (K-pad 150..159)
#define C1 30
#define CSP 32            // 30 real pooled channels + 2 zero (K-pad 270..287)
#define C2 75
#define HO 128
#define HP 130            // pooled + pad(1) each side
#define THR1 15.0f
#define THR2 10.0f
#define NHW (128*128)     // 16384
#define NCHW (75*16384)   // 1228800 per timestep
#define K1 160            // conv1 K padded (6*5*5=150 -> 160)
#define K2 288            // conv2 K padded (30*3*3=270 -> 288)
#define K1P 264           // LDS row stride (halfs): 132 words == 4 mod 64 -> conflict-free b128
#define K2P 392           // LDS row stride (halfs): 196 words == 4 mod 64 -> conflict-free b128
static constexpr size_t NPOT = (size_t)TT * C2 * HO * HO;   // 18,432,000

// ---------------- workspace layout (bytes) ----------------
static constexpr size_t OFF_XP     = 0;                          // f16 [15,7,260,260]
static constexpr size_t OFF_S      = 14196224;                   // f16 [15,32,130,130]
static constexpr size_t SZ_S       = (size_t)TT*CSP*HP*HP*2;     // 16,224,000
static constexpr size_t OFF_ZEND   = OFF_S + SZ_S;               // 30,420,224 (zeroed range end)
static constexpr size_t OFF_BP1    = 30420224;                   // conv1 B frags: 2*5*32*16 halfs
static constexpr size_t OFF_BP2    = OFF_BP1 + 10240;
static constexpr size_t OFF_WINMAP = OFF_BP2 + 46080;            // int [16384]
static constexpr size_t OFF_COUNT  = OFF_WINMAP + 65536;         // int [1228800]
static constexpr size_t OFF_VALS   = OFF_COUNT + 4915200;        // float [1228800]
static constexpr size_t OFF_GMAX   = OFF_VALS + 4915200;         // int [1]
static constexpr size_t OFF_SUPF   = OFF_GMAX + 256;             // int [75]
static constexpr size_t OFF_WINR   = OFF_SUPF + 512;             // int [8]
static constexpr size_t OFF_WINC   = OFF_WINR + 64;              // int [8]
static constexpr size_t OFF_PVAL   = OFF_WINC + 64;              // float [240]
static constexpr size_t OFF_PIDX   = OFF_PVAL + 1024;            // int [240]

// K-chunk element index for WMMA f16 A/B fragments (16 halfs per lane):
// lane<16 holds K = {hs*8+0..7, 16+hs*8+0..7} with hs = lane>>4.
__device__ __forceinline__ int frag_k(int kk, int hs, int j) {
    return kk * 32 + ((j < 8) ? (hs * 8 + j) : (16 + hs * 8 + (j - 8)));
}

// ---------------- init: zero Xp + S (contiguous) + small state ----------------
__global__ void k_init(unsigned int* __restrict__ wsWords, int* __restrict__ gmax,
                       int* __restrict__ supF) {
    size_t i = (size_t)blockIdx.x * blockDim.x + threadIdx.x;
    const size_t n = OFF_ZEND / 4;
    const size_t stride = (size_t)gridDim.x * blockDim.x;
    for (size_t j = i; j < n; j += stride) wsWords[j] = 0u;
    if (i == 0) *gmax = 0;
    if (i < C2) supF[i] = 0;
}

// ---------------- pre-pad: x f32 -> f16 into Xp with pad=2 baked in ----------------
__global__ void k_prepad(const float* __restrict__ x, _Float16* __restrict__ Xp) {
    size_t i = (size_t)blockIdx.x * blockDim.x + threadIdx.x;
    const size_t n = (size_t)TT * CIN * HIN * HIN;
    const size_t stride = (size_t)gridDim.x * blockDim.x;
    for (; i < n; i += stride) {
        int ix = (int)(i & 255);
        int iy = (int)((i >> 8) & 255);
        int c  = (int)((i >> 16) % CIN);
        int t  = (int)(i / ((size_t)CIN * HIN * HIN));
        Xp[(((size_t)t * CXP + c) * HXP + (iy + 2)) * HXP + (ix + 2)] = (_Float16)x[i];
    }
}

// ---------------- pack weights into WMMA B-fragment layout (f16) ----------------
__global__ void k_pack(const float* __restrict__ w1, const float* __restrict__ w2a,
                       _Float16* __restrict__ bp1, _Float16* __restrict__ bp2) {
    int tid = blockIdx.x * blockDim.x + threadIdx.x;
    if (tid < 2 * 5 * 32) {                       // conv1: nt(2) x kk(5) x lane(32)
        int lane = tid & 31, frag = tid >> 5;
        int kk = frag % 5, nt = frag / 5;
        int n = nt * 16 + (lane & 15), hs = lane >> 4;
        _Float16* dst = bp1 + (size_t)tid * 16;
        #pragma unroll
        for (int j = 0; j < 16; ++j) {
            int k = frag_k(kk, hs, j);
            float v = 0.0f;
            if (n < C1 && k < 150) {              // k = c*25 + ky*5 + kx
                int c = k / 25, r = k % 25, ky = r / 5, kx = r % 5;
                v = w1[(((size_t)n * CIN + c) * 5 + ky) * 5 + kx];
            }
            dst[j] = (_Float16)v;
        }
    } else if (tid < 2 * 5 * 32 + 5 * 9 * 32) {   // conv2: nt(5) x kk(9) x lane(32)
        int t2 = tid - 2 * 5 * 32;
        int lane = t2 & 31, frag = t2 >> 5;
        int kk = frag % 9, nt = frag / 9;
        int n = nt * 16 + (lane & 15), hs = lane >> 4;
        _Float16* dst = bp2 + (size_t)t2 * 16;
        #pragma unroll
        for (int j = 0; j < 16; ++j) {
            int k = frag_k(kk, hs, j);
            float v = 0.0f;
            if (n < C2 && k < 270) {              // k = c*9 + ky*3 + kx
                int c = k / 9, r = k % 9, ky = r / 3, kx = r % 3;
                v = w2a[(((size_t)n * C1 + c) * 3 + ky) * 3 + kx];
            }
            dst[j] = (_Float16)v;
        }
    }
}

// ---------------- conv1 + fire + 2x2 maxpool -> padded pooled f16 spikes ----------------
// Block = 128 threads (4 waves) = 64 output cols of one pool row; im2col tile staged in LDS.
// Fragment loads become 2x ds_load_b128 per K-chunk (offsets linear in k).
__global__ void __launch_bounds__(128) k_conv1(const _Float16* __restrict__ Xp,
                                               const _Float16* __restrict__ bp1,
                                               _Float16* __restrict__ S) {
    __shared__ _Float16 smem[64 * K1P];           // 33,792 B
    const int tid  = threadIdx.x;
    const int wid  = tid >> 5;
    const int lane = tid & 31;
    const int qa   = blockIdx.x & 3;              // 64-col quarter
    const int py   = (blockIdx.x >> 2) & 127;     // pooled row
    const int t    = blockIdx.x >> 9;
    const int ox0  = qa * 64;
    const int mrow = lane & 15, hs = lane >> 4;
    const int ox_base = ox0 + wid * 16;
    const _Float16* Xt = Xp + (size_t)t * (CXP * HXP * HXP);

    const int mf = tid & 63, khalf = tid >> 6;    // fill coords (k wave-uniform)
    const _Float16* lbase = &smem[(size_t)(wid * 16 + mrow) * K1P + hs * 8];

    float pm[2][4];
    #pragma unroll
    for (int nt = 0; nt < 2; ++nt)
        #pragma unroll
        for (int p = 0; p < 4; ++p) pm[nt][p] = -1e30f;

    for (int row = 0; row < 2; ++row) {
        const int oy = 2 * py + row;              // padded row index of ky=0 tap
        // cooperative coalesced im2col fill: fixed k -> 64 contiguous source halfs
        for (int k = khalf; k < K1; k += 2) {
            int c = k / 25, r = k % 25, ky = r / 5, kx = r % 5;
            smem[(size_t)mf * K1P + k] =
                Xt[((size_t)c * HXP + (oy + ky)) * HXP + (ox0 + mf + kx)];
        }
        __syncthreads();

        v16h A[5];
        #pragma unroll
        for (int kk = 0; kk < 5; ++kk) {          // two ds_load_b128 per chunk
            V16J u;
            u.h[0] = *(const v8h*)(lbase + kk * 32);
            u.h[1] = *(const v8h*)(lbase + kk * 32 + 16);
            A[kk] = u.v;
        }
        #pragma unroll
        for (int nt = 0; nt < 2; ++nt) {
            v8f acc = {};
            #pragma unroll
            for (int kk = 0; kk < 5; ++kk) {
                v16h b = *(const v16h*)(bp1 + (((size_t)nt * 5 + kk) * 32 + lane) * 16);
                acc = __builtin_amdgcn_wmma_f32_16x16x32_f16(
                          false, A[kk], false, b, (short)0, acc, false, false);
            }
            #pragma unroll
            for (int p = 0; p < 4; ++p)
                pm[nt][p] = fmaxf(pm[nt][p], fmaxf(acc[2 * p], acc[2 * p + 1]));
        }
        __syncthreads();                           // LDS reuse for second row
    }
    // pooled spike = any of the 2x2 potentials >= THR1; write with pad offset (+1,+1)
    #pragma unroll
    for (int nt = 0; nt < 2; ++nt) {
        int ch = nt * 16 + (lane & 15);
        if (ch < C1) {                             // channels 30/31 stay zero (K-pad)
            size_t base = (((size_t)t * CSP + ch) * HP + (1 + py)) * HP
                        + 1 + (ox_base >> 1) + hs * 4;
            #pragma unroll
            for (int p = 0; p < 4; ++p)
                S[base + p] = (pm[nt][p] >= THR1) ? (_Float16)1.0f : (_Float16)0.0f;
        }
    }
}

// ---------------- conv2 + fire -> truncated potentials (d_out pot region) ----------------
// Block = 128 threads (4 waves) = 64 output cols of one row; im2col tile staged in LDS.
__global__ void __launch_bounds__(128) k_conv2(const _Float16* __restrict__ S,
                                               const _Float16* __restrict__ bp2,
                                               float* __restrict__ P) {
    __shared__ _Float16 smem[64 * K2P];           // 50,176 B
    const int tid  = threadIdx.x;
    const int wid  = tid >> 5;
    const int lane = tid & 31;
    const int half = blockIdx.x & 1;              // 64-col half of the row
    const int oy   = (blockIdx.x >> 1) & 127;
    const int t    = blockIdx.x >> 8;
    const int ox0  = half * 64;
    const int mrow = lane & 15, hs = lane >> 4;
    const int ox_base = ox0 + wid * 16;
    const _Float16* St = S + (size_t)t * (CSP * HP * HP);

    const int mf = tid & 63, khalf = tid >> 6;
    for (int k = khalf; k < K2; k += 2) {         // k wave-uniform -> scalar addr math
        int c = k / 9, r = k % 9, ky = r / 3, kx = r % 3;
        smem[(size_t)mf * K2P + k] =
            St[((size_t)c * HP + (oy + ky)) * HP + (ox0 + mf + kx)];
    }
    __syncthreads();

    const _Float16* lbase = &smem[(size_t)(wid * 16 + mrow) * K2P + hs * 8];
    v16h A[9];
    #pragma unroll
    for (int kk = 0; kk < 9; ++kk) {              // two ds_load_b128 per chunk
        V16J u;
        u.h[0] = *(const v8h*)(lbase + kk * 32);
        u.h[1] = *(const v8h*)(lbase + kk * 32 + 16);
        A[kk] = u.v;
    }
    #pragma unroll
    for (int nt = 0; nt < 5; ++nt) {
        v8f acc = {};
        #pragma unroll
        for (int kk = 0; kk < 9; ++kk) {
            v16h b = *(const v16h*)(bp2 + (((size_t)nt * 9 + kk) * 32 + lane) * 16);
            acc = __builtin_amdgcn_wmma_f32_16x16x32_f16(
                      false, A[kk], false, b, (short)0, acc, false, false);
        }
        int ch = nt * 16 + (lane & 15);
        if (ch < C2) {
            size_t base = (((size_t)t * C2 + ch) * HO + oy) * HO + ox_base + hs * 8;
            float o[8];
            #pragma unroll
            for (int v = 0; v < 8; ++v) o[v] = (acc[v] < THR2) ? 0.0f : acc[v];
            *(float4*)(P + base)     = make_float4(o[0], o[1], o[2], o[3]);
            *(float4*)(P + base + 4) = make_float4(o[4], o[5], o[6], o[7]);
        }
    }
}

// ---------------- pointwise inhibition: winner map per (h,w) ----------------
__global__ void k_inhib_map(const float* __restrict__ P, int* __restrict__ winmap) {
    int hw = blockIdx.x * blockDim.x + threadIdx.x;
    if (hw >= NHW) return;
    int cnt = 0; float mxLast = 0.0f;
    for (int t = 0; t < TT; ++t) {
        float mx = -1.0f;
        for (int c = 0; c < C2; ++c)
            mx = fmaxf(mx, P[((size_t)t * C2 + c) * NHW + hw]);
        if (mx > 0.0f) ++cnt;
        if (t == TT - 1) mxLast = mx;
    }
    int e = TT - cnt; e = e < 0 ? 0 : (e > TT - 1 ? TT - 1 : e);
    float best = -1.0f; int bi = 0;                 // first-max argmax over c
    for (int c = 0; c < C2; ++c) {
        float v = P[((size_t)e * C2 + c) * NHW + hw];
        if (v > best) { best = v; bi = c; }
    }
    winmap[hw] = (mxLast > 0.0f) ? bi : -1;         // -1 == never spiked (coef 0)
}

// ---------------- apply inhibition in place, emit spikes ----------------
__global__ void k_inhib_apply(float* __restrict__ P, float* __restrict__ SPK,
                              const int* __restrict__ winmap) {
    size_t i = (size_t)blockIdx.x * blockDim.x + threadIdx.x;
    const size_t stride = (size_t)gridDim.x * blockDim.x;
    for (; i < NPOT; i += stride) {
        int hw = (int)(i % NHW);
        int c  = (int)((i / NHW) % C2);
        float v = P[i];
        v = (c == winmap[hw]) ? v : 0.0f;
        P[i] = v;
        SPK[i] = (v > 0.0f) ? 1.0f : 0.0f;
    }
}

// ---------------- kWTA stats: spike count, value at earliest, global max ----------------
__global__ void k_kw_stats(const float* __restrict__ P, int* __restrict__ cntA,
                           float* __restrict__ valA, int* __restrict__ gmax) {
    int i = blockIdx.x * blockDim.x + threadIdx.x;
    if (i >= NCHW) return;
    int cnt = 0;
    for (int t = 0; t < TT; ++t) cnt += (P[(size_t)t * NCHW + i] > 0.0f) ? 1 : 0;
    int e = TT - cnt; e = e < 0 ? 0 : (e > TT - 1 ? TT - 1 : e);
    float val = P[(size_t)e * NCHW + i];
    cntA[i] = cnt; valA[i] = val;
    if (cnt > 0) atomicMax(gmax, __float_as_int(val));   // val >= 0 -> bit-order == value-order
}

// ---------------- kWTA iteration: partial first-max argmax with suppression ----------------
__global__ void k_kw_scan(const int* __restrict__ cntA, const float* __restrict__ valA,
                          const int* __restrict__ gmax, const int* __restrict__ supF,
                          const int* __restrict__ winR, const int* __restrict__ winC,
                          float* __restrict__ pval, int* __restrict__ pidx, int it) {
    const float v = __int_as_float(*gmax) * (float)TT;
    float bval = -1.0f; int bidx = 0;
    const int tid = blockIdx.x * blockDim.x + threadIdx.x;
    const int stride = gridDim.x * blockDim.x;
    for (int i = tid; i < NCHW; i += stride) {
        int c = i >> 14, hw = i & 16383;
        int r = hw >> 7, cc = hw & 127;
        float tot = 0.0f;
        int cnt = cntA[i];
        if (cnt > 0 && supF[c] == 0) {
            bool sup = false;
            for (int j = 0; j < it; ++j) {
                int dr = r - winR[j]; if (dr < 0) dr = -dr;
                int dc = cc - winC[j]; if (dc < 0) dc = -dc;
                sup = sup || (dr <= 1 && dc <= 1);
            }
            if (!sup) tot = (float)cnt * (valA[i] + v);
        }
        if (tot > bval) { bval = tot; bidx = i; }   // strict '>' keeps first index
    }
    __shared__ float sv[256]; __shared__ int si[256];
    sv[threadIdx.x] = bval; si[threadIdx.x] = bidx;
    __syncthreads();
    for (int s = 128; s > 0; s >>= 1) {
        if ((int)threadIdx.x < s) {
            if (sv[threadIdx.x + s] > sv[threadIdx.x] ||
                (sv[threadIdx.x + s] == sv[threadIdx.x] &&
                 si[threadIdx.x + s] < si[threadIdx.x])) {
                sv[threadIdx.x] = sv[threadIdx.x + s];
                si[threadIdx.x] = si[threadIdx.x + s];
            }
        }
        __syncthreads();
    }
    if (threadIdx.x == 0) { pval[blockIdx.x] = sv[0]; pidx[blockIdx.x] = si[0]; }
}

__global__ void k_kw_pick(const float* __restrict__ pval, const int* __restrict__ pidx,
                          int* __restrict__ supF, int* __restrict__ winR,
                          int* __restrict__ winC, float* __restrict__ outw,
                          int it, int nparts) {
    __shared__ float sv[256]; __shared__ int si[256];
    float bval = -1.0f; int bidx = 0;
    if ((int)threadIdx.x < nparts) { bval = pval[threadIdx.x]; bidx = pidx[threadIdx.x]; }
    sv[threadIdx.x] = bval; si[threadIdx.x] = bidx;
    __syncthreads();
    for (int s = 128; s > 0; s >>= 1) {
        if ((int)threadIdx.x < s) {
            if (sv[threadIdx.x + s] > sv[threadIdx.x] ||
                (sv[threadIdx.x + s] == sv[threadIdx.x] &&
                 si[threadIdx.x + s] < si[threadIdx.x])) {
                sv[threadIdx.x] = sv[threadIdx.x + s];
                si[threadIdx.x] = si[threadIdx.x + s];
            }
        }
        __syncthreads();
    }
    if (threadIdx.x == 0) {
        int idx = si[0]; float val = sv[0];
        int f = idx >> 14, hw = idx & 16383, r = hw >> 7, c = hw & 127;
        bool valid = val > 0.0f;
        outw[it * 3 + 0] = valid ? (float)f : -1.0f;
        outw[it * 3 + 1] = valid ? (float)r : -1.0f;
        outw[it * 3 + 2] = valid ? (float)c : -1.0f;
        supF[f] = 1;                  // reference suppresses with argmax result either way
        winR[it] = r; winC[it] = c;
    }
}

extern "C" void kernel_launch(void* const* d_in, const int* in_sizes, int n_in,
                              void* d_out, int out_size, void* d_ws, size_t ws_size,
                              hipStream_t stream) {
    (void)in_sizes; (void)n_in; (void)out_size; (void)ws_size;
    const float* x   = (const float*)d_in[0];
    // d_in[1]=max_layer, d_in[2]=layer_branch: fixed training path per reference
    const float* w1  = (const float*)d_in[3];
    const float* w2a = (const float*)d_in[4];

    float* out = (float*)d_out;
    float* SPK = out;                 // spk_a  [15,75,128,128]
    float* P   = out + NPOT;          // pot_a  [15,75,128,128] (built in place)
    float* WIN = out + 2 * NPOT;      // winners [8,3] as floats

    char* ws = (char*)d_ws;
    _Float16* Xp  = (_Float16*)(ws + OFF_XP);
    _Float16* S   = (_Float16*)(ws + OFF_S);
    _Float16* BP1 = (_Float16*)(ws + OFF_BP1);
    _Float16* BP2 = (_Float16*)(ws + OFF_BP2);
    int*   winmap = (int*)  (ws + OFF_WINMAP);
    int*   cntA   = (int*)  (ws + OFF_COUNT);
    float* valA   = (float*)(ws + OFF_VALS);
    int*   gmax   = (int*)  (ws + OFF_GMAX);
    int*   supF   = (int*)  (ws + OFF_SUPF);
    int*   winR   = (int*)  (ws + OFF_WINR);
    int*   winC   = (int*)  (ws + OFF_WINC);
    float* pval   = (float*)(ws + OFF_PVAL);
    int*   pidx   = (int*)  (ws + OFF_PIDX);

    k_init  <<<2048, 256, 0, stream>>>((unsigned int*)ws, gmax, supF);
    k_prepad<<<4096, 256, 0, stream>>>(x, Xp);
    k_pack  <<<(2*5*32 + 5*9*32 + 255) / 256, 256, 0, stream>>>(w1, w2a, BP1, BP2);
    k_conv1 <<<TT * 128 * 4, 128, 0, stream>>>(Xp, BP1, S);
    k_conv2 <<<TT * 128 * 2, 128, 0, stream>>>(S, BP2, P);
    k_inhib_map  <<<(NHW + 255) / 256, 256, 0, stream>>>(P, winmap);
    k_inhib_apply<<<4096, 256, 0, stream>>>(P, SPK, winmap);
    k_kw_stats   <<<(NCHW + 255) / 256, 256, 0, stream>>>(P, cntA, valA, gmax);
    for (int it = 0; it < 8; ++it) {
        k_kw_scan<<<240, 256, 0, stream>>>(cntA, valA, gmax, supF, winR, winC, pval, pidx, it);
        k_kw_pick<<<1, 256, 0, stream>>>(pval, pidx, supF, winR, winC, WIN, it, 240);
    }
}
